// MemoryEfficientEdgeConvBlock_2216203125200
// MI455X (gfx1250) — matrix operand order repaired and checked by reading, not compiled, not run on hardware
//
#include <hip/hip_runtime.h>
#include <hip/hip_bf16.h>

typedef _Float16 v16h __attribute__((ext_vector_type(16)));
typedef _Float16 v8h  __attribute__((ext_vector_type(8)));
typedef float    v8f  __attribute__((ext_vector_type(8)));

#define NPTS 32768
#define CFEAT 128
#define OUTF 128
#define KNN 16

// workspace layout (bytes)
#define XH_OFF   (0)                                  // N*128 f16 = 8 MB
#define ZY_OFF   (XH_OFF + NPTS * 128 * 2)            // N*256 f16 = 16 MB
#define IDX_OFF  (ZY_OFF + NPTS * 256 * 2)            // N*16 i32 = 2 MB
#define WABT_OFF (IDX_OFF + NPTS * 16 * 4)            // 256*128 f16 = 64 KB
#define W2T_OFF  (WABT_OFF + 256 * 128 * 2)           // 128*128 f16 = 32 KB
#define POS4_OFF (W2T_OFF + 128 * 128 * 2)            // N*float4 = 512 KB

// ---------------------------------------------------------------------------
// Weight prep + pos packing.
// WabT[n][c]: n<128 -> Wa^T = W1[c][n]-W1[c+128][n]; n>=128 -> Wb^T.
// W2T[p][o] = W2[o][p].
// pos4[j] = (-2x, -2y, -2z, |p|^2)  so knn distance is a pure 3-FMA chain.
// ---------------------------------------------------------------------------
__global__ void prep_weights(const float* __restrict__ W1,
                             const float* __restrict__ W2,
                             const float* __restrict__ pos,
                             _Float16* __restrict__ WabT,
                             _Float16* __restrict__ W2T,
                             float4* __restrict__ pos4) {
    int t = blockIdx.x * blockDim.x + threadIdx.x;
    int stride = gridDim.x * blockDim.x;
    for (int i = t; i < 256 * 128; i += stride) {
        int n = i >> 7, c = i & 127;
        float v = (n < 128) ? (W1[c * 128 + n] - W1[(c + 128) * 128 + n])
                            : W1[(c + 128) * 128 + (n - 128)];
        WabT[n * 128 + c] = (_Float16)v;
    }
    for (int i = t; i < 128 * 128; i += stride) {
        int p = i >> 7, o = i & 127;
        W2T[p * 128 + o] = (_Float16)W2[o * 128 + p];
    }
    for (int i = t; i < NPTS; i += stride) {
        float px = pos[i * 3 + 0], py = pos[i * 3 + 1], pz = pos[i * 3 + 2];
        pos4[i] = make_float4(-2.0f * px, -2.0f * py, -2.0f * pz,
                              px * px + py * py + pz * pz);
    }
}

// ---------------------------------------------------------------------------
// x (f32) -> xh (f16)
// ---------------------------------------------------------------------------
__global__ void convert_x(const float* __restrict__ x, _Float16* __restrict__ xh,
                          int total) {
    int t = blockIdx.x * blockDim.x + threadIdx.x;
    for (int i = t; i < total; i += gridDim.x * blockDim.x)
        xh[i] = (_Float16)x[i];
}

// ---------------------------------------------------------------------------
// Exact kNN: one thread per query, float4-packed LDS candidate tiles
// (one broadcast ds_load_b128 + 3 FMA + 1 cmp per candidate), register top-16.
// d = |p_j|^2 - 2 q.p_j  (same ranking as reference).
// ---------------------------------------------------------------------------
__global__ __launch_bounds__(256) void knn_kernel(const float* __restrict__ pos,
                                                  const float4* __restrict__ pos4,
                                                  int* __restrict__ idxOut) {
    __shared__ float4 sp[256];
    int tid = threadIdx.x;
    int q = blockIdx.x * 256 + tid;
    float qx = pos[q * 3 + 0], qy = pos[q * 3 + 1], qz = pos[q * 3 + 2];

    float dist[16];
    int ind[16];
#pragma unroll
    for (int t = 0; t < 16; ++t) { dist[t] = 3.4e38f; ind[t] = 0; }

    for (int tile = 0; tile < NPTS; tile += 256) {
        __syncthreads();
        sp[tid] = pos4[tile + tid];
        __syncthreads();
#pragma unroll 4
        for (int c = 0; c < 256; ++c) {
            float4 s = sp[c];
            float d = fmaf(qz, s.z, fmaf(qy, s.y, fmaf(qx, s.x, s.w)));
            int jj = tile + c;
            if (jj != q && d < dist[15]) {
                dist[15] = d; ind[15] = jj;
#pragma unroll
                for (int t = 15; t > 0; --t) {
                    if (dist[t] < dist[t - 1]) {
                        float td = dist[t]; dist[t] = dist[t - 1]; dist[t - 1] = td;
                        int ti = ind[t];   ind[t]  = ind[t - 1];  ind[t - 1]  = ti;
                    }
                }
            }
        }
    }
#pragma unroll
    for (int t = 0; t < 16; ++t) idxOut[q * 16 + t] = ind[t];
}

// ---------------------------------------------------------------------------
// GEMM1: zy[N][256] (f16) = xh[N][128] @ Wab[128][256]
// One wave per 16(M) x 32(N): A fragments loaded once, shared across two
// N-tiles -> 8 WMMAs per 24 global_load_b128 (halves redundant A traffic).
// ---------------------------------------------------------------------------
__global__ __launch_bounds__(256) void gemm1(const _Float16* __restrict__ xh,
                                             const _Float16* __restrict__ WabT,
                                             _Float16* __restrict__ zy) {
    int lane = threadIdx.x & 31;
    int wid = threadIdx.x >> 5;
    int gw = blockIdx.x * 8 + wid;
    int mTile = gw >> 3;   // 0..2047
    int nPair = gw & 7;    // owns N-tiles {2*nPair, 2*nPair+1}
    int hi = lane >> 4;    // half-wave select

    const _Float16* arow = xh + (mTile * 16 + (lane & 15)) * 128;

    // A 16x32 f16 layout: lane(l%16)=row; elems 0..15 = K hi*16+0..15
    v16h a[4];
#pragma unroll
    for (int k = 0; k < 4; ++k) {
        v8h alo = *(const v8h*)(arow + k * 32 + hi * 16);
        v8h ahi = *(const v8h*)(arow + k * 32 + hi * 16 + 8);
#pragma unroll
        for (int i = 0; i < 8; ++i) { a[k][i] = alo[i]; a[k][8 + i] = ahi[i]; }
    }

#pragma unroll
    for (int n = 0; n < 2; ++n) {
        int nTile = nPair * 2 + n;
        const _Float16* bcol = WabT + (nTile * 16 + (lane & 15)) * 128;
        v8f c = {};
#pragma unroll
        for (int k = 0; k < 4; ++k) {
            // B 32x16 f16: lane(l%16)=col; elems = K hi*16+0..15 (contig in W^T)
            v16h b = *(const v16h*)(bcol + k * 32 + hi * 16);
            c = __builtin_amdgcn_wmma_f32_16x16x32_f16(false, a[k], false, b,
                                                       (short)0, c, false, false);
        }
        // C layout: VGPR r, lane l: row = r + 8*(l>=16), col = l%16
        int ccol = nTile * 16 + (lane & 15);
#pragma unroll
        for (int r = 0; r < 8; ++r)
            zy[(mTile * 16 + r + hi * 8) * 256 + ccol] = (_Float16)c[r];
    }
}

// ---------------------------------------------------------------------------
// Fused edge kernel: block = 16 points. Two passes of 8 neighbors:
//   H[128][128] = relu(z_i + y_j + b1) built in LDS (f16, padded stride),
//   H @ W2 via WMMA (8 M-tiles x 2 N-tiles per wave x 4 K-steps),
//   neighbor-max folded into the C-register layout, + b2 after max.
// ---------------------------------------------------------------------------
__global__ __launch_bounds__(128) void edge_fused(const _Float16* __restrict__ zy,
                                                  const int* __restrict__ nidx,
                                                  const _Float16* __restrict__ W2T,
                                                  const float* __restrict__ b1,
                                                  const float* __restrict__ b2,
                                                  float* __restrict__ out) {
    __shared__ _Float16 Hs[128][136];   // stride 136 f16 = 272B (16B multiple)
    __shared__ int sIdx[256];
    __shared__ float sB1[128];

    int tid = threadIdx.x;
    int lane = tid & 31;
    int w = tid >> 5;         // wave 0..3, owns N-tiles {2w, 2w+1}
    int hi = lane >> 4;
    int base = blockIdx.x * 16;

    for (int i = tid; i < 256; i += 128) sIdx[i] = nidx[base * 16 + i];
    sB1[tid] = b1[tid];

    // B fragments of W2 (kept in registers across both passes)
    v16h bfrag[2][4];
#pragma unroll
    for (int n = 0; n < 2; ++n) {
        const _Float16* bp = W2T + ((w * 2 + n) * 16 + (lane & 15)) * 128;
#pragma unroll
        for (int k = 0; k < 4; ++k)
            bfrag[n][k] = *(const v16h*)(bp + k * 32 + hi * 16);
    }

    float rmax[8][2];
#pragma unroll
    for (int m = 0; m < 8; ++m) { rmax[m][0] = -3.4e38f; rmax[m][1] = -3.4e38f; }

    for (int kp = 0; kp < 2; ++kp) {
        __syncthreads();   // idx/b1 ready (kp=0); previous GEMM reads done (kp=1)
        // build 128 edge rows: row e -> point e>>3, neighbor kp*8 + (e&7)
        {
            int e = tid;
            int p = e >> 3;
            int j = sIdx[p * 16 + kp * 8 + (e & 7)];
            const _Float16* zr = zy + (base + p) * 256;        // z_i part
            const _Float16* yr = zy + j * 256 + 128;           // y_j part
#pragma unroll
            for (int c8 = 0; c8 < 16; ++c8) {
                v8h zv = *(const v8h*)(zr + c8 * 8);
                v8h yv = *(const v8h*)(yr + c8 * 8);
                v8h hv;
#pragma unroll
                for (int i = 0; i < 8; ++i) {
                    float v = (float)zv[i] + (float)yv[i] + sB1[c8 * 8 + i];
                    hv[i] = (_Float16)(v > 0.0f ? v : 0.0f);
                }
                *(v8h*)(&Hs[e][c8 * 8]) = hv;
            }
        }
        __syncthreads();

#pragma unroll
        for (int m = 0; m < 8; ++m) {
            const _Float16* ar = &Hs[m * 16 + (lane & 15)][0];
            v16h a[4];
#pragma unroll
            for (int k = 0; k < 4; ++k) {
                v8h lo = *(const v8h*)(ar + k * 32 + hi * 16);
                v8h hh = *(const v8h*)(ar + k * 32 + hi * 16 + 8);
#pragma unroll
                for (int i = 0; i < 8; ++i) { a[k][i] = lo[i]; a[k][8 + i] = hh[i]; }
            }
#pragma unroll
            for (int n = 0; n < 2; ++n) {
                v8f c = {};
#pragma unroll
                for (int k = 0; k < 4; ++k)
                    c = __builtin_amdgcn_wmma_f32_16x16x32_f16(false, a[k], false,
                                                               bfrag[n][k],
                                                               (short)0, c,
                                                               false, false);
                // rows r=0..7 of this tile belong to point 2m+hi -> per-lane max
                float mx = c[0];
#pragma unroll
                for (int r = 1; r < 8; ++r) mx = fmaxf(mx, c[r]);
                rmax[m][n] = fmaxf(rmax[m][n], mx);
            }
        }
    }

#pragma unroll
    for (int m = 0; m < 8; ++m) {
        int p = m * 2 + hi;
#pragma unroll
        for (int n = 0; n < 2; ++n) {
            int col = (w * 2 + n) * 16 + (lane & 15);
            out[(base + p) * 128 + col] = rmax[m][n] + b2[col];
        }
    }
}

extern "C" void kernel_launch(void* const* d_in, const int* in_sizes, int n_in,
                              void* d_out, int out_size, void* d_ws, size_t ws_size,
                              hipStream_t stream) {
    const float* x   = (const float*)d_in[0];
    const float* pos = (const float*)d_in[1];
    const float* W1  = (const float*)d_in[2];
    const float* b1  = (const float*)d_in[3];
    const float* W2  = (const float*)d_in[4];
    const float* b2  = (const float*)d_in[5];
    float* out = (float*)d_out;

    char* ws = (char*)d_ws;
    _Float16* xh   = (_Float16*)(ws + XH_OFF);
    _Float16* zy   = (_Float16*)(ws + ZY_OFF);
    int*      nidx = (int*)(ws + IDX_OFF);
    _Float16* WabT = (_Float16*)(ws + WABT_OFF);
    _Float16* W2T  = (_Float16*)(ws + W2T_OFF);
    float4*   pos4 = (float4*)(ws + POS4_OFF);

    prep_weights<<<64, 256, 0, stream>>>(W1, W2, pos, WabT, W2T, pos4);
    convert_x<<<4096, 256, 0, stream>>>(x, xh, NPTS * 128);
    knn_kernel<<<NPTS / 256, 256, 0, stream>>>(pos, pos4, nidx);
    gemm1<<<2048, 256, 0, stream>>>(xh, WabT, zy);
    edge_fused<<<NPTS / 16, 128, 0, stream>>>(zy, nidx, W2T, b1, b2, out);
}
